// AGCRN_2954937499759
// MI455X (gfx1250) — matrix operand order repaired and checked by reading, not compiled, not run on hardware
//
#include <hip/hip_runtime.h>

typedef unsigned short u16;
typedef __attribute__((ext_vector_type(16))) __bf16 v16bf;
typedef __attribute__((ext_vector_type(8)))  float  v8f;

#define NN   2048   // nodes
#define BB   64     // batch
#define DD   16     // embedding
#define CC   128    // concat channels (in+out)
#define OO   64     // out channels

// ---------- bf16 helpers ----------
__device__ __forceinline__ u16 f2bf(float x) {
  union { float f; unsigned u; } v; v.f = x;
  unsigned r = v.u + 0x7FFFu + ((v.u >> 16) & 1u);
  return (u16)(r >> 16);
}
__device__ __forceinline__ float bf2f(u16 h) {
  union { unsigned u; float f; } v; v.u = ((unsigned)h) << 16; return v.f;
}

struct BFPK { uint4 lo, hi; };
// A-operand: per-lane K = kh*8 + {0..7} and 16 + kh*8 + {0..7}  (ISA 16-bit A 16x32 layout)
__device__ __forceinline__ v16bf load_a_op(const u16* p) {
  BFPK u;
  u.lo = *reinterpret_cast<const uint4*>(p);
  u.hi = *reinterpret_cast<const uint4*>(p + 16);
  return __builtin_bit_cast(v16bf, u);
}
// B-operand: per-lane K = kh*8 + {0..7} and 16 + kh*8 + {0..7} (same as A)
__device__ __forceinline__ v16bf load_b_op(const u16* p) {
  BFPK u;
  u.lo = *reinterpret_cast<const uint4*>(p);
  u.hi = *reinterpret_cast<const uint4*>(p + 16);
  return __builtin_bit_cast(v16bf, u);
}
__device__ __forceinline__ v8f wmma_bf16(v16bf a, v16bf b, v8f c) {
  return __builtin_amdgcn_wmma_f32_16x16x32_bf16(false, a, false, b, (short)0, c,
                                                 false, false);
}

// ---------- kernel 1: A = softmax(relu(E E^T)) row-wise, stored bf16 ----------
__global__ __launch_bounds__(256)
void softmax_rows(const float* __restrict__ E, u16* __restrict__ Abf) {
  const int n = blockIdx.x;
  __shared__ float row[NN];
  __shared__ float red[256];
  float e[DD];
#pragma unroll
  for (int d = 0; d < DD; ++d) e[d] = E[n * DD + d];

  float lmax = 0.0f;  // relu output >= 0
  for (int m = threadIdx.x; m < NN; m += 256) {
    float s = 0.f;
#pragma unroll
    for (int d = 0; d < DD; ++d) s = fmaf(e[d], E[m * DD + d], s);
    s = fmaxf(s, 0.f);
    row[m] = s;
    lmax = fmaxf(lmax, s);
  }
  red[threadIdx.x] = lmax; __syncthreads();
  for (int off = 128; off > 0; off >>= 1) {
    if (threadIdx.x < off) red[threadIdx.x] = fmaxf(red[threadIdx.x], red[threadIdx.x + off]);
    __syncthreads();
  }
  const float mx = red[0]; __syncthreads();

  float lsum = 0.f;
  for (int m = threadIdx.x; m < NN; m += 256) {
    float v = __expf(row[m] - mx);
    row[m] = v;
    lsum += v;
  }
  red[threadIdx.x] = lsum; __syncthreads();
  for (int off = 128; off > 0; off >>= 1) {
    if (threadIdx.x < off) red[threadIdx.x] += red[threadIdx.x + off];
    __syncthreads();
  }
  const float inv = 1.0f / red[0];
  for (int m = threadIdx.x; m < NN; m += 256)
    Abf[(size_t)n * NN + m] = f2bf(row[m] * inv);
}

// ---------- kernel 2: pack Vg = bf16([X|H]), Vu[:, :64] = bf16(X) ----------
__global__ __launch_bounds__(256)
void pack_feats(const float* __restrict__ X, const float* __restrict__ H,
                u16* __restrict__ Vg, u16* __restrict__ Vu) {
  const size_t i = (size_t)blockIdx.x * 256 + threadIdx.x;
  if (i >= (size_t)BB * NN * OO) return;
  const int c = (int)(i & 63);
  const size_t bn = i >> 6;
  const u16 xb = f2bf(X[i]);
  const u16 hb = f2bf(H[i]);
  Vg[bn * CC + c]      = xb;
  Vg[bn * CC + 64 + c] = hb;
  Vu[bn * CC + c]      = xb;
}

// ---------- kernel 3/4: Y = A@V (MODE 0) or Y = 2*A@V - U (MODE 1), bf16 out ----------
// Block tile: M=128, N=128, K-step=32. Double-buffered LDS, 1 barrier/iter.
// Each wave owns 2 M-tiles x 4 N-tiles: every B reg load feeds 2 WMMAs,
// every A reg load feeds 4 WMMAs, acc stays at 64 VGPRs (no spills).
template <int MODE>
__global__ __launch_bounds__(256)
void gemm_AV(const u16* __restrict__ Abf, const u16* __restrict__ V,
             const u16* __restrict__ U, u16* __restrict__ Y) {
  const int m0 = blockIdx.x * 128;          // node-row tile
  const int b  = blockIdx.y;                // batch
  const u16* Vb = V + (size_t)b * NN * CC;
  u16* Yb       = Y + (size_t)b * NN * CC;

  __shared__ __align__(16) u16 As[2][128 * 32];   // [m][k]
  __shared__ __align__(16) u16 Vs[2][128 * 32];   // [f][k] (transposed)

  const int tid  = threadIdx.x;
  const int w    = tid >> 5;
  const int lane = tid & 31;
  const int lr   = lane & 15;
  const int kh   = lane >> 4;
  const int mgroup = w & 3;    // rows mgroup*32 + {0..31}
  const int ngroup = w >> 2;   // cols ngroup*64 + nt*16

  // staging maps
  const int arow = tid >> 1, acb = (tid & 1) * 16;  // A: 16 bf16 per thread
  const int vk2  = (tid >> 4) * 2;                  // V: 2 k-rows x 8 f per thread
  const int vfb  = (tid & 15) * 8;

  v8f acc[2][4];
#pragma unroll
  for (int mi = 0; mi < 2; ++mi)
#pragma unroll
    for (int i = 0; i < 4; ++i)
#pragma unroll
      for (int j = 0; j < 8; ++j) acc[mi][i][j] = 0.f;

  uint4 ra0, ra1;    // 16 bf16 of one A half-row
  uint4 rv0, rv1;    // 8 bf16 of two V rows

  auto stage_load = [&](int kt) {
    const int kbase = kt * 32;
    const uint4* pa = reinterpret_cast<const uint4*>(
        Abf + (size_t)(m0 + arow) * NN + kbase + acb);
    ra0 = pa[0]; ra1 = pa[1];
    rv0 = *reinterpret_cast<const uint4*>(Vb + (size_t)(kbase + vk2) * CC + vfb);
    rv1 = *reinterpret_cast<const uint4*>(Vb + (size_t)(kbase + vk2 + 1) * CC + vfb);
  };
  auto stage_store = [&](int buf) {
    uint4* da = reinterpret_cast<uint4*>(&As[buf][arow * 32 + acb]);
    da[0] = ra0; da[1] = ra1;
    u16 t0[8], t1[8];
    *reinterpret_cast<uint4*>(t0) = rv0;
    *reinterpret_cast<uint4*>(t1) = rv1;
#pragma unroll
    for (int j = 0; j < 8; ++j) {
      const unsigned pk = (unsigned)t0[j] | ((unsigned)t1[j] << 16);
      *reinterpret_cast<unsigned*>(&Vs[buf][(vfb + j) * 32 + vk2]) = pk;
    }
  };

  stage_load(0);
  stage_store(0);
  __syncthreads();

  for (int kt = 0; kt < NN / 32; ++kt) {
    const int cur = kt & 1;
    if (kt < NN / 32 - 1) stage_load(kt + 1);   // global loads fly over the WMMA chain

    const v16bf a0 = load_a_op(&As[cur][(mgroup * 32 + lr) * 32 + kh * 8]);
    const v16bf a1 = load_a_op(&As[cur][(mgroup * 32 + 16 + lr) * 32 + kh * 8]);
#pragma unroll
    for (int nt = 0; nt < 4; ++nt) {
      const v16bf bm = load_b_op(&Vs[cur][(ngroup * 64 + nt * 16 + lr) * 32 + kh * 8]);
      acc[0][nt] = wmma_bf16(a0, bm, acc[0][nt]);
      acc[1][nt] = wmma_bf16(a1, bm, acc[1][nt]);
    }

    if (kt < NN / 32 - 1) stage_store(cur ^ 1);
    __syncthreads();
  }

  // epilogue: C/D layout -> row m = base + v + 8*kh, col = ngroup*64 + nt*16 + lr
#pragma unroll
  for (int mi = 0; mi < 2; ++mi)
#pragma unroll
    for (int nt = 0; nt < 4; ++nt) {
      const int f = ngroup * 64 + nt * 16 + lr;
#pragma unroll
      for (int v = 0; v < 8; ++v) {
        const int m = m0 + mgroup * 32 + mi * 16 + v + 8 * kh;
        float val = acc[mi][nt][v];
        if (MODE == 1)
          val = 2.f * val - bf2f(U[((size_t)b * NN + m) * CC + f]);
        Yb[(size_t)m * CC + f] = f2bf(val);
      }
    }
}

// ---------- kernel 5/8: per-node [B x 3C] @ W_n[3C x O], fused GRU epilogues ----------
// PHASE 0 (O=128): ZR=sigmoid(.); write R, and Vu[:,64:]=bf16(Z*H)
// PHASE 1 (O=64):  HC=tanh(.); out = R*H + (1-R)*HC
template <int O, int PHASE>
__global__ __launch_bounds__(256)
void node_mm(const u16* __restrict__ F0, const u16* __restrict__ F1,
             const u16* __restrict__ F2, const float* __restrict__ E,
             const float* __restrict__ pool, const float* __restrict__ bias,
             const float* __restrict__ H, float* __restrict__ Rbuf,
             u16* __restrict__ Vu, float* __restrict__ out) {
  constexpr int NT  = O / 16;   // N-tiles
  constexpr int NTW = NT / 2;   // N-tiles per wave
  constexpr int OB  = O / 8;    // o-columns per staging thread

  const int n = blockIdx.x;
  __shared__ __align__(16) u16 XGs[64 * 32];   // [batch][k']
  __shared__ __align__(16) u16 Ws[O * 32];     // [o][k'] (transposed)
  __shared__ float Es[DD];

  const int tid  = threadIdx.x;
  const int w    = tid >> 5;
  const int lane = tid & 31;
  const int lr   = lane & 15;
  const int kh   = lane >> 4;
  const int mtile = w & 3;
  const int ntb   = (w >> 2) * NTW;

  if (tid < DD) Es[tid] = E[n * DD + tid];
  __syncthreads();

  v8f acc[NTW];
#pragma unroll
  for (int i = 0; i < NTW; ++i)
#pragma unroll
    for (int j = 0; j < 8; ++j) acc[i][j] = 0.f;

  for (int c = 0; c < 12; ++c) {
    const int k  = c >> 2;          // Chebyshev order
    const int i0 = (c & 3) * 32;    // channel base
    const u16* Fk = (k == 0) ? F0 : (k == 1) ? F1 : F2;

    {  // stage XG tile: [64 batches][32 channels]
      const int bt = tid >> 2, ib = (tid & 3) * 8;
      const uint4* src = reinterpret_cast<const uint4*>(
          Fk + ((size_t)bt * NN + n) * CC + i0 + ib);
      *reinterpret_cast<uint4*>(&XGs[bt * 32 + ib]) = *src;
    }
    {  // compute W_n chunk = sum_d E[n,d] * pool[d,k,i0+kp, ob..ob+OB) -> Ws transposed
      const int kp = tid >> 3, ob = (tid & 7) * OB;
      float wacc[OB];
#pragma unroll
      for (int j = 0; j < OB; ++j) wacc[j] = 0.f;
#pragma unroll
      for (int d = 0; d < DD; ++d) {
        const float ev = Es[d];
        const float* ps = pool + (((size_t)d * 3 + k) * CC + (i0 + kp)) * O + ob;
#pragma unroll
        for (int q = 0; q < OB / 4; ++q) {
          const float4 pv = reinterpret_cast<const float4*>(ps)[q];
          wacc[q * 4 + 0] = fmaf(ev, pv.x, wacc[q * 4 + 0]);
          wacc[q * 4 + 1] = fmaf(ev, pv.y, wacc[q * 4 + 1]);
          wacc[q * 4 + 2] = fmaf(ev, pv.z, wacc[q * 4 + 2]);
          wacc[q * 4 + 3] = fmaf(ev, pv.w, wacc[q * 4 + 3]);
        }
      }
#pragma unroll
      for (int j = 0; j < OB; ++j) Ws[(ob + j) * 32 + kp] = f2bf(wacc[j]);
    }
    __syncthreads();

    const v16bf a = load_a_op(&XGs[(mtile * 16 + lr) * 32 + kh * 8]);
#pragma unroll
    for (int ntl = 0; ntl < NTW; ++ntl) {
      const v16bf bmat = load_b_op(&Ws[((ntb + ntl) * 16 + lr) * 32 + kh * 8]);
      acc[ntl] = wmma_bf16(a, bmat, acc[ntl]);
    }
    __syncthreads();
  }

  // fused epilogues
#pragma unroll
  for (int ntl = 0; ntl < NTW; ++ntl) {
    const int o = (ntb + ntl) * 16 + lr;
    float bv = 0.f;
#pragma unroll
    for (int d = 0; d < DD; ++d) bv = fmaf(Es[d], bias[d * O + o], bv);
#pragma unroll
    for (int v = 0; v < 8; ++v) {
      const int brow = mtile * 16 + v + 8 * kh;
      const size_t bn = (size_t)brow * NN + n;
      const float x = acc[ntl][v] + bv;
      if (PHASE == 0) {
        const float zr = 1.0f / (1.0f + __expf(-x));
        if (o < OO) {  // Z half -> Vu[:, 64:] = bf16(Z * H)
          const float h = H[bn * OO + o];
          Vu[bn * CC + OO + o] = f2bf(zr * h);
        } else {       // R half
          Rbuf[bn * OO + (o - OO)] = zr;
        }
      } else {
        const float hc = tanhf(x);
        const float r  = Rbuf[bn * OO + o];
        const float h  = H[bn * OO + o];
        out[bn * OO + o] = r * h + (1.0f - r) * hc;
      }
    }
  }
}

// ---------- host launch ----------
extern "C" void kernel_launch(void* const* d_in, const int* in_sizes, int n_in,
                              void* d_out, int out_size, void* d_ws, size_t ws_size,
                              hipStream_t stream) {
  const float* X  = (const float*)d_in[0];
  const float* E  = (const float*)d_in[1];
  const float* H  = (const float*)d_in[2];
  const float* gW = (const float*)d_in[3];
  const float* gB = (const float*)d_in[4];
  const float* uW = (const float*)d_in[5];
  const float* uB = (const float*)d_in[6];
  float* out = (float*)d_out;

  char* ws = (char*)d_ws;
  const size_t A_BYTES = (size_t)NN * NN * 2;        // 8 MiB, L2-resident
  const size_t F_BYTES = (size_t)BB * NN * CC * 2;   // 32 MiB each
  u16*   Abf = (u16*)(ws);
  u16*   Vg  = (u16*)(ws + A_BYTES);
  u16*   Y1  = (u16*)(ws + A_BYTES + 1 * F_BYTES);
  u16*   Y2  = (u16*)(ws + A_BYTES + 2 * F_BYTES);
  u16*   Vu  = (u16*)(ws + A_BYTES + 3 * F_BYTES);
  float* Rb  = (float*)(ws + A_BYTES + 4 * F_BYTES);

  // 1) adjacency
  softmax_rows<<<NN, 256, 0, stream>>>(E, Abf);
  // 2) pack features
  const int packBlocks = (BB * NN * OO) / 256;
  pack_feats<<<packBlocks, 256, 0, stream>>>(X, H, Vg, Vu);

  const dim3 ggrid(NN / 128, BB);
  // 3) gate Chebyshev: Y1 = A@Vg ; Y2 = 2*A@Y1 - Vg
  gemm_AV<0><<<ggrid, 256, 0, stream>>>(Abf, Vg, nullptr, Y1);
  gemm_AV<1><<<ggrid, 256, 0, stream>>>(Abf, Y1, Vg, Y2);
  // 4) gate node matmul: sigmoid, emit R and Vu[:,64:] = bf16(Z*H)
  node_mm<128, 0><<<NN, 256, 0, stream>>>(Vg, Y1, Y2, E, gW, gB, H, Rb, Vu, nullptr);
  // 5) update Chebyshev on Vu
  gemm_AV<0><<<ggrid, 256, 0, stream>>>(Abf, Vu, nullptr, Y1);
  gemm_AV<1><<<ggrid, 256, 0, stream>>>(Abf, Y1, Vu, Y2);
  // 6) update node matmul: tanh + final GRU blend into d_out
  node_mm<64, 1><<<NN, 256, 0, stream>>>(Vu, Y1, Y2, E, uW, uB, H, Rb, nullptr, out);

  (void)in_sizes; (void)n_in; (void)out_size; (void)ws_size;
}
